// SoftMLPMoEBlock_24111946400255
// MI455X (gfx1250) — matrix operand order repaired and checked by reading, not compiled, not run on hardware
//
#include <hip/hip_runtime.h>
#include <hip/hip_bf16.h>
#include <math.h>

// ---------------- problem constants ----------------
#define BB   4
#define NN   4096
#define DD   1024
#define EE   16
#define SS   64
#define FF   2048
#define DOUT 1024
#define ESL  (EE*SS)      // 1024 slots total
#define MTOK (BB*NN)      // 16384 tokens
#define LNEPS 1e-5f

// ---------------- WMMA GEMM tiling ----------------
#define BM 128
#define BN 128
#define TK 32
#define LDSK 40           // TK + 8 pad (avoid LDS bank conflicts)
#define GEMM_THREADS 256  // 8 wave32 waves

typedef __attribute__((ext_vector_type(16))) __bf16 v16bf;
typedef __attribute__((ext_vector_type(8)))  float  v8f;
typedef int v4i_vs __attribute__((vector_size(16)));   // builtin's b128 unit

// ---- CDNA5 async global->LDS copy (ASYNCcnt-tracked), probe-guarded ----
#if defined(__AMDGCN__) && defined(__has_builtin)
#  if __has_builtin(__builtin_amdgcn_global_load_async_to_lds_b128) && \
      __has_builtin(__builtin_amdgcn_s_wait_asynccnt)
#    define USE_ASYNC_COPY 1
#  endif
#endif
#ifndef USE_ASYNC_COPY
#  define USE_ASYNC_COPY 0
#endif

__device__ __forceinline__ void copy16_g2lds(const __bf16* g, __bf16* l) {
#if USE_ASYNC_COPY
    __builtin_amdgcn_global_load_async_to_lds_b128(
        (__attribute__((address_space(1))) v4i_vs*)(g),
        (__attribute__((address_space(3))) v4i_vs*)(l), 0, 0);
#else
    *(uint4*)l = *(const uint4*)g;
#endif
}

__device__ __forceinline__ void copy_fence() {
#if USE_ASYNC_COPY
    __builtin_amdgcn_s_wait_asynccnt(0);
#endif
}

__device__ __forceinline__ float gelu_exact(float x) {
    return 0.5f * x * (1.0f + erff(x * 0.70710678118654752f));
}

// ============================================================
// Block-tiled bf16 WMMA GEMM:  C[M,N'] = A[M,K] * B (+bias, act)
//   BKMAJOR : B stored [K,N'] row-major (weights/xn) vs [N',K] (se^T)
//   OUTBF16 : store bf16 intermediate vs f32
//   ACT=1   : exact GELU after bias
//   HASB/HASM: compile-time bias / output-row-remap presence
// 128x128 block tile, 8 waves, wave tile 32x64 = 2x4 WMMA tiles,
// double-buffered LDS with async (or vector) staging of next K-slab.
// Dims are multiples of BM/BN/TK for every call here.
// ============================================================
template<int BKMAJOR, int OUTBF16, int ACT, int HASB, int HASM>
__global__ __launch_bounds__(GEMM_THREADS)
void gemm_wmma_bf16(const __bf16* __restrict__ A,
                    const __bf16* __restrict__ Bm,
                    void* __restrict__ Cv,
                    int K, int lda, int ldb, int ldc,
                    long long strideA, long long strideB, long long strideC,
                    const float* __restrict__ bias, int biasStride,
                    const int* __restrict__ mapC, int mapStride)
{
    __shared__ __bf16 As[2][BM * LDSK];
    __shared__ __bf16 Bs[2][BN * LDSK];

    const int z = blockIdx.z;
    A  += (size_t)z * strideA;
    Bm += (size_t)z * strideB;

    const int tid   = threadIdx.x;
    const int lane  = tid & 31;
    const int wave  = tid >> 5;        // 0..7
    const int waveM = wave & 3;        // 4 waves along M -> 32 rows each
    const int waveN = wave >> 2;       // 2 waves along N -> 64 cols each
    const int mBase = blockIdx.y * BM;
    const int nBase = blockIdx.x * BN;
    const int kh    = lane >> 4;       // lane half (0/1)
    const int lr    = lane & 15;

    auto stage = [&](int buf, int k0) {
        // A tile: BM x TK, 16B chunks, coalesced (async when available)
#pragma unroll
        for (int i = 0; i < (BM * TK / 8) / GEMM_THREADS; ++i) {   // 2
            int idx = tid + i * GEMM_THREADS;
            int row = idx >> 2, ch = idx & 3;
            copy16_g2lds(A + (size_t)(mBase + row) * lda + k0 + ch * 8,
                         &As[buf][row * LDSK + ch * 8]);
        }
        if (BKMAJOR) {
            // B [K,N']: vector b128 global reads, bf16 scatter into Bs[n][k]
#pragma unroll
            for (int i = 0; i < (BN * TK / 8) / GEMM_THREADS; ++i) { // 2
                int idx = tid + i * GEMM_THREADS;
                int kk = idx >> 4, nc = (idx & 15) * 8;             // 16 vec/row
                union { uint4 q; __bf16 e[8]; } u;
                u.q = *(const uint4*)(Bm + (size_t)(k0 + kk) * ldb + nBase + nc);
#pragma unroll
                for (int j = 0; j < 8; ++j)
                    Bs[buf][(nc + j) * LDSK + kk] = u.e[j];
            }
        } else {
            // B [N',K]: same pattern as A
#pragma unroll
            for (int i = 0; i < (BN * TK / 8) / GEMM_THREADS; ++i) { // 2
                int idx = tid + i * GEMM_THREADS;
                int row = idx >> 2, ch = idx & 3;
                copy16_g2lds(Bm + (size_t)(nBase + row) * ldb + k0 + ch * 8,
                             &Bs[buf][row * LDSK + ch * 8]);
            }
        }
    };

    v8f acc[2][4];
#pragma unroll
    for (int i = 0; i < 2; ++i)
#pragma unroll
        for (int j = 0; j < 4; ++j)
#pragma unroll
            for (int q = 0; q < 8; ++q) acc[i][j][q] = 0.0f;

    const int nk = K / TK;
    stage(0, 0);
    for (int it = 0; it < nk; ++it) {
        const int cur = it & 1;
        copy_fence();                 // s_wait_asynccnt 0 (async path)
        __syncthreads();              // staged tile visible; prev reads done
        if (it + 1 < nk) stage(cur ^ 1, (it + 1) * TK);

        // ---- fragments (ISA 16-bit A/B VGPR layouts, wave32) ----
        v16bf aF[2], bF[4];
#pragma unroll
        for (int t = 0; t < 2; ++t) {
            // A 16x32: lane half -> K base 0/8; elems 8..15 = K+16..23
            const __bf16* pa = &As[cur][(waveM * 32 + t * 16 + lr) * LDSK + kh * 8];
            union { v16bf v; uint4 q[2]; } ua;
            ua.q[0] = *(const uint4*)pa;
            ua.q[1] = *(const uint4*)(pa + 16);
            aF[t] = ua.v;
        }
#pragma unroll
        for (int t = 0; t < 4; ++t) {
            // B 32x16: lane half -> K base 0/8; elems 8..15 = K+16..23
            const __bf16* pb = &Bs[cur][(waveN * 64 + t * 16 + lr) * LDSK + kh * 8];
            union { v16bf v; uint4 q[2]; } ub;
            ub.q[0] = *(const uint4*)pb;
            ub.q[1] = *(const uint4*)(pb + 16);
            bF[t] = ub.v;
        }
#pragma unroll
        for (int i = 0; i < 2; ++i)
#pragma unroll
            for (int j = 0; j < 4; ++j)
                acc[i][j] = __builtin_amdgcn_wmma_f32_16x16x32_bf16(
                    false, aF[i], false, bF[j], (short)0, acc[i][j], false, false);
    }

    // ---- epilogue: C VGPR q -> (m = 8*half + q, n = lane&15) ----
#pragma unroll
    for (int i = 0; i < 2; ++i) {
#pragma unroll
        for (int j = 0; j < 4; ++j) {
            int nGlob = nBase + waveN * 64 + j * 16 + lr;
            float bv = HASB ? bias[(size_t)z * biasStride + nGlob] : 0.0f;
#pragma unroll
            for (int q = 0; q < 8; ++q) {
                int mIn  = mBase + waveM * 32 + i * 16 + kh * 8 + q;
                int orow = HASM ? mapC[(size_t)z * mapStride + mIn] : mIn;
                float val = acc[i][j][q] + bv;
                if (ACT == 1) val = gelu_exact(val);
                size_t cidx = (size_t)z * strideC + (size_t)orow * ldc + nGlob;
                if (OUTBF16) ((__bf16*)Cv)[cidx] = (__bf16)val;
                else         ((float*)Cv)[cidx]  = val;
            }
        }
    }
}

// ============================================================
// LayerNorm over rows of length DD -> bf16 (optional *scale)
// ============================================================
__global__ __launch_bounds__(256)
void layernorm_bf16(const float* __restrict__ x, const float* __restrict__ g,
                    const float* __restrict__ bsh, const float* __restrict__ scale,
                    __bf16* __restrict__ out)
{
    __shared__ float red[256];
    const int t = threadIdx.x;
    const size_t row = blockIdx.x;
    const float* xr = x + row * DD;
    float v[4]; float s = 0.0f;
#pragma unroll
    for (int i = 0; i < 4; ++i) { v[i] = xr[t + i * 256]; s += v[i]; }
    red[t] = s; __syncthreads();
    for (int o = 128; o > 0; o >>= 1) { if (t < o) red[t] += red[t + o]; __syncthreads(); }
    float mu = red[0] * (1.0f / DD);
    __syncthreads();
    float qq = 0.0f;
#pragma unroll
    for (int i = 0; i < 4; ++i) { float d = v[i] - mu; qq += d * d; }
    red[t] = qq; __syncthreads();
    for (int o = 128; o > 0; o >>= 1) { if (t < o) red[t] += red[t + o]; __syncthreads(); }
    float rstd = rsqrtf(red[0] * (1.0f / DD) + LNEPS);
    float sc = scale ? scale[0] : 1.0f;
#pragma unroll
    for (int i = 0; i < 4; ++i) {
        int c = t + i * 256;
        out[row * DD + c] = (__bf16)(((v[i] - mu) * rstd * g[c] + bsh[c]) * sc);
    }
}

// ============================================================
// dispatch softmax: over tokens (column of logits), transposed bf16 store
// ============================================================
__global__ __launch_bounds__(256)
void softmax_dispatch(const float* __restrict__ logits, __bf16* __restrict__ dispT)
{
    __shared__ float red[256];
    const int t  = threadIdx.x;
    const int b  = blockIdx.x >> 10;
    const int es = blockIdx.x & (ESL - 1);
    const float* base = logits + (size_t)b * NN * ESL + es;
    float v[16]; float mx = -3.4e38f;
#pragma unroll
    for (int i = 0; i < 16; ++i) { v[i] = base[(size_t)(t + i * 256) * ESL]; mx = fmaxf(mx, v[i]); }
    red[t] = mx; __syncthreads();
    for (int o = 128; o > 0; o >>= 1) { if (t < o) red[t] = fmaxf(red[t], red[t + o]); __syncthreads(); }
    float gm = red[0]; __syncthreads();
    float s = 0.0f;
#pragma unroll
    for (int i = 0; i < 16; ++i) { v[i] = __expf(v[i] - gm); s += v[i]; }
    red[t] = s; __syncthreads();
    for (int o = 128; o > 0; o >>= 1) { if (t < o) red[t] += red[t + o]; __syncthreads(); }
    float inv = 1.0f / red[0];
    __bf16* o2 = dispT + ((size_t)b * ESL + es) * NN;
#pragma unroll
    for (int i = 0; i < 16; ++i) o2[t + i * 256] = (__bf16)(v[i] * inv);
}

// combine softmax: per-token row of ESL
__global__ __launch_bounds__(256)
void softmax_combine(const float* __restrict__ logits, __bf16* __restrict__ comb)
{
    __shared__ float red[256];
    const int t = threadIdx.x;
    const float* base = logits + (size_t)blockIdx.x * ESL;
    float v[4]; float mx = -3.4e38f;
#pragma unroll
    for (int i = 0; i < 4; ++i) { v[i] = base[t + i * 256]; mx = fmaxf(mx, v[i]); }
    red[t] = mx; __syncthreads();
    for (int o = 128; o > 0; o >>= 1) { if (t < o) red[t] = fmaxf(red[t], red[t + o]); __syncthreads(); }
    float gm = red[0]; __syncthreads();
    float s = 0.0f;
#pragma unroll
    for (int i = 0; i < 4; ++i) { v[i] = __expf(v[i] - gm); s += v[i]; }
    red[t] = s; __syncthreads();
    for (int o = 128; o > 0; o >>= 1) { if (t < o) red[t] += red[t + o]; __syncthreads(); }
    float inv = 1.0f / red[0];
    __bf16* o2 = comb + (size_t)blockIdx.x * ESL;
#pragma unroll
    for (int i = 0; i < 4; ++i) o2[t + i * 256] = (__bf16)(v[i] * inv);
}

__global__ void f32_to_bf16(const float* __restrict__ in, __bf16* __restrict__ out, long long n)
{
    long long i = (long long)blockIdx.x * blockDim.x + threadIdx.x;
    long long st = (long long)gridDim.x * blockDim.x;
    for (; i < n; i += st) out[i] = (__bf16)in[i];
}

__global__ void gate_mul(const float* __restrict__ g, const float* __restrict__ u,
                         __bf16* __restrict__ h, long long n)
{
    long long i = (long long)blockIdx.x * blockDim.x + threadIdx.x;
    long long st = (long long)gridDim.x * blockDim.x;
    for (; i < n; i += st) h[i] = (__bf16)(g[i] * u[i]);
}

// row-remap tables: slots GEMM out [b, e*S+s] -> [e, b*S+s]; down GEMM out -> [b, e*S+s]
__global__ void init_maps(int* __restrict__ sMap, int* __restrict__ dMap)
{
    int idx = blockIdx.x * 256 + threadIdx.x;
    if (idx < BB * ESL) {
        int b = idx >> 10, m = idx & (ESL - 1), e = m >> 6, s = m & 63;
        sMap[idx] = e * (BB * SS) + b * SS + s;
    }
    if (idx < EE * BB * SS) {
        int e = idx >> 8, m = idx & 255, b = m >> 6, s = m & 63;
        dMap[idx] = b * ESL + e * SS + s;
    }
}

// ============================================================
extern "C" void kernel_launch(void* const* d_in, const int* in_sizes, int n_in,
                              void* d_out, int out_size, void* d_ws, size_t ws_size,
                              hipStream_t stream)
{
    (void)in_sizes; (void)n_in; (void)out_size; (void)ws_size;
    const float* x       = (const float*)d_in[0];
    const float* norm_g  = (const float*)d_in[1];
    const float* norm_b  = (const float*)d_in[2];
    const float* slot_g  = (const float*)d_in[3];
    const float* slot_b  = (const float*)d_in[4];
    const float* scale   = (const float*)d_in[5];
    const float* sembeds = (const float*)d_in[6];
    const float* Wg      = (const float*)d_in[7];
    const float* bg      = (const float*)d_in[8];
    const float* Wu      = (const float*)d_in[9];
    const float* bu      = (const float*)d_in[10];
    const float* Wd      = (const float*)d_in[11];
    const float* bd      = (const float*)d_in[12];
    float* out = (float*)d_out;

    char* ws = (char*)d_ws;
    size_t off = 0;
    auto take = [&](size_t bytes) -> char* {
        char* p = ws + off; off += (bytes + 255) & ~(size_t)255; return p;
    };
    __bf16* xn     = (__bf16*)take((size_t)MTOK * DD * 2);
    __bf16* se     = (__bf16*)take((size_t)ESL * DD * 2);
    float*  logits = (float*) take((size_t)MTOK * ESL * 4);
    __bf16* dispT  = (__bf16*)take((size_t)BB * ESL * NN * 2);
    __bf16* comb   = (__bf16*)take((size_t)MTOK * ESL * 2);
    __bf16* slots  = (__bf16*)take((size_t)EE * BB * SS * DD * 2);
    __bf16* WgB    = (__bf16*)take((size_t)EE * DD * FF * 2);
    __bf16* WuB    = (__bf16*)take((size_t)EE * DD * FF * 2);
    __bf16* WdB    = (__bf16*)take((size_t)EE * FF * DOUT * 2);
    __bf16* eout   = (__bf16*)take((size_t)BB * ESL * DOUT * 2);
    int* slotsMap  = (int*)take((size_t)BB * ESL * 4);
    int* downMap   = (int*)take((size_t)EE * BB * SS * 4);
    // reuse: logits region -> gate/up f32 after softmaxes; dispT -> h after slots GEMM
    float*  gateF = logits;
    float*  upF   = logits + (size_t)EE * BB * SS * FF;
    __bf16* hB    = dispT;

    init_maps<<<16, 256, 0, stream>>>(slotsMap, downMap);
    layernorm_bf16<<<MTOK, 256, 0, stream>>>(x, norm_g, norm_b, nullptr, xn);
    layernorm_bf16<<<ESL, 256, 0, stream>>>(sembeds, slot_g, slot_b, scale, se);
    f32_to_bf16<<<8192, 256, 0, stream>>>(Wg, WgB, (long long)EE * DD * FF);
    f32_to_bf16<<<8192, 256, 0, stream>>>(Wu, WuB, (long long)EE * DD * FF);
    f32_to_bf16<<<8192, 256, 0, stream>>>(Wd, WdB, (long long)EE * FF * DOUT);

    // logits = xn [16384,1024] * se^T   (B is N-major [ESL, DD])
    gemm_wmma_bf16<0, 0, 0, 0, 0><<<dim3(ESL / BN, MTOK / BM, 1), GEMM_THREADS, 0, stream>>>(
        xn, se, logits, DD, DD, DD, ESL, 0, 0, 0, nullptr, 0, nullptr, 0);

    softmax_dispatch<<<BB * ESL, 256, 0, stream>>>(logits, dispT);
    softmax_combine<<<MTOK, 256, 0, stream>>>(logits, comb);

    // slots[e, b*S+s, :] = dispT[b] [ESL,NN] * xn[b] [NN,DD]   (batched over b)
    gemm_wmma_bf16<1, 1, 0, 0, 1><<<dim3(DD / BN, ESL / BM, BB), GEMM_THREADS, 0, stream>>>(
        dispT, xn, slots, NN, NN, DD, DD,
        (long long)ESL * NN, (long long)NN * DD, 0,
        nullptr, 0, slotsMap, ESL);

    // gate = gelu(slots[e] * Wg[e] + bg[e])   (batched over e)
    gemm_wmma_bf16<1, 0, 1, 1, 0><<<dim3(FF / BN, (BB * SS) / BM, EE), GEMM_THREADS, 0, stream>>>(
        slots, WgB, gateF, DD, DD, FF, FF,
        (long long)BB * SS * DD, (long long)DD * FF, (long long)BB * SS * FF,
        bg, FF, nullptr, 0);
    // up = slots[e] * Wu[e] + bu[e]
    gemm_wmma_bf16<1, 0, 0, 1, 0><<<dim3(FF / BN, (BB * SS) / BM, EE), GEMM_THREADS, 0, stream>>>(
        slots, WuB, upF, DD, DD, FF, FF,
        (long long)BB * SS * DD, (long long)DD * FF, (long long)BB * SS * FF,
        bu, FF, nullptr, 0);
    gate_mul<<<8192, 256, 0, stream>>>(gateF, upF, hB, (long long)EE * BB * SS * FF);

    // eout[b, e*S+s, :] = h[e] * Wd[e] + bd[e]
    gemm_wmma_bf16<1, 1, 0, 1, 1><<<dim3(DOUT / BN, (BB * SS) / BM, EE), GEMM_THREADS, 0, stream>>>(
        hB, WdB, eout, FF, FF, DOUT, DOUT,
        (long long)BB * SS * FF, (long long)FF * DOUT, 0,
        bd, DOUT, downMap, BB * SS);

    // out[b] = comb[b] [NN,ESL] * eout[b] [ESL,DOUT]
    gemm_wmma_bf16<1, 0, 0, 0, 0><<<dim3(DOUT / BN, NN / BM, BB), GEMM_THREADS, 0, stream>>>(
        comb, eout, out, ESL, ESL, DOUT, DOUT,
        (long long)NN * ESL, (long long)ESL * DOUT, (long long)NN * DOUT,
        nullptr, 0, nullptr, 0);
}
